// Decoder_50190987821448
// MI455X (gfx1250) — matrix operand order repaired and checked by reading, not compiled, not run on hardware
//
#include <hip/hip_runtime.h>
#include <hip/hip_bf16.h>
#include <math.h>

// ---------------------------------------------------------------------------
// Types for CDNA5 WMMA / TDM
// ---------------------------------------------------------------------------
typedef __attribute__((ext_vector_type(16))) __bf16  v16bf;
typedef __attribute__((ext_vector_type(8)))  float   v8f;
typedef __attribute__((ext_vector_type(4)))  unsigned int u32x4;
typedef __attribute__((ext_vector_type(8)))  int     i32x8;
typedef __attribute__((ext_vector_type(4)))  int     i32x4;

#define HIDDEN 512
#define NB_GRAPHS 64

// ---------------------------------------------------------------------------
// TDM: DMA a 2D tile (rows x 32 fp32, row stride ldelem) from global -> LDS.
// D# packing per CDNA5 ISA ch.8.3/8.4: group0 = {count/flags, lds_addr,
// global_addr, type=2}; group1 = {data_size=4B, tensor dims, tile dims,
// strides}. 2D tile: groups 2/3 unused (tile_dim2=0), passed as zeros.
// This toolchain's builtin is the 6-arg form (extra int32x8, unused here).
// Issued once per wave (EXEC ignored); caller gates to one wave and
// synchronizes with s_wait_tensorcnt + workgroup barrier.
// ---------------------------------------------------------------------------
__device__ __forceinline__ void tdm_load_tile(unsigned lds_addr, const float* gptr,
                                              unsigned rows, unsigned ldelem)
{
    unsigned long long ga = (unsigned long long)(uintptr_t)gptr;
    u32x4 g0;
    g0[0] = 1u;                                            // count=1, user mode
    g0[1] = lds_addr;                                      // LDS byte address
    g0[2] = (unsigned)(ga & 0xFFFFFFFFu);                  // global_addr[31:0]
    g0[3] = (unsigned)((ga >> 32) & 0x01FFFFFFu)           // global_addr[56:32]
            | 0x80000000u;                                 // type=2 ("image")

    const unsigned td0 = 32u;          // tensor_dim0 (x extent for OOB)
    const unsigned td1 = rows;         // tensor_dim1
    const unsigned long long s0 = (unsigned long long)ldelem; // dim0 stride
    i32x8 g1;
    g1[0] = (int)(2u << 16);                               // data_size=2 -> 4B
    g1[1] = (int)((td0 & 0xFFFFu) << 16);                  // tensor_dim0[15:0]
    g1[2] = (int)((td0 >> 16) | ((td1 & 0xFFFFu) << 16));  // td0 hi | td1 lo
    g1[3] = (int)((td1 >> 16) | (32u << 16));              // td1 hi | tile_dim0=32
    g1[4] = (int)rows;                                     // tile_dim1 | tile_dim2=0
    g1[5] = (int)(s0 & 0xFFFFFFFFu);                       // dim0_stride lo
    g1[6] = (int)((s0 >> 32) & 0xFFFFu);                   // dim0_stride hi
    g1[7] = 0;
    i32x4 z4 = {0, 0, 0, 0};
    i32x8 z8 = {0, 0, 0, 0, 0, 0, 0, 0};
    __builtin_amdgcn_tensor_load_to_lds(g0, g1, z4, z4, z8, 0);
}

// ---------------------------------------------------------------------------
// Big GEMM with TDM staging:  C[M,N] = A[M,K] @ W[N,K]^T + bias
// Block tile 128x64, 8 waves, double-buffered fp32 LDS tiles filled by the
// Tensor Data Mover; fp32->bf16 conversion at fragment build (co-executes
// with XDL WMMA); fp32 accumulation via v_wmma_f32_16x16x32_bf16.
// ---------------------------------------------------------------------------
__global__ __launch_bounds__(256) void wmma_gemm_tdm_kernel(
    const float* __restrict__ A, const float* __restrict__ W, int K, int ldw,
    const float* __restrict__ bias, float* __restrict__ C, int N)
{
    constexpr int BM = 128, BN = 64;
    __shared__ float Asf[2][BM][32];   // 2 x 16 KB
    __shared__ float Bsf[2][BN][32];   // 2 x  8 KB

    const int tid  = threadIdx.x;
    const int wave = tid >> 5;
    const int half = (tid >> 4) & 1;
    const int l16  = tid & 15;

    const int row0 = blockIdx.x * BM;
    const int col0 = blockIdx.y * BN;

    v8f acc[4];
#pragma unroll
    for (int j = 0; j < 4; ++j)
#pragma unroll
        for (int g = 0; g < 8; ++g) acc[j][g] = 0.0f;

    const int nchunks = K >> 5;

    // prologue: wave 0 issues TDM for chunk 0 into buffer 0
    if (wave == 0) {
        tdm_load_tile((unsigned)(uintptr_t)&Asf[0][0][0],
                      A + (size_t)row0 * K, BM, (unsigned)K);
        tdm_load_tile((unsigned)(uintptr_t)&Bsf[0][0][0],
                      W + (size_t)col0 * ldw, BN, (unsigned)ldw);
    }

    int buf = 0;
    for (int kc = 0; kc < nchunks; ++kc) {
        if (wave == 0) __builtin_amdgcn_s_wait_tensorcnt(0);  // chunk kc landed
        __syncthreads();                                      // publish buffer
        if (wave == 0 && kc + 1 < nchunks) {                  // prefetch kc+1
            const int k1 = (kc + 1) << 5;
            tdm_load_tile((unsigned)(uintptr_t)&Asf[buf ^ 1][0][0],
                          A + (size_t)row0 * K + k1, BM, (unsigned)K);
            tdm_load_tile((unsigned)(uintptr_t)&Bsf[buf ^ 1][0][0],
                          W + (size_t)col0 * ldw + k1, BN, (unsigned)ldw);
        }

        // A fragment (16x32 bf16): lane l16 = row; layout:
        //   elem e -> k = half*16+e (matches B)
        v16bf a;
        {
            const float* ar = &Asf[buf][wave * 16 + l16][0];
#pragma unroll
            for (int e = 0; e < 8; ++e) {
                a[e]     = (__bf16)ar[half * 16 + e];
                a[e + 8] = (__bf16)ar[half * 16 + 8 + e];
            }
        }
        // B fragments (32x16 bf16): lane l16 = n; k = half*16 + e
        v16bf bf[4];
#pragma unroll
        for (int j = 0; j < 4; ++j) {
            const float* br = &Bsf[buf][j * 16 + l16][0];
#pragma unroll
            for (int e = 0; e < 16; ++e) bf[j][e] = (__bf16)br[half * 16 + e];
        }
#pragma unroll
        for (int j = 0; j < 4; ++j)
            acc[j] = __builtin_amdgcn_wmma_f32_16x16x32_bf16(
                false, a, false, bf[j], (short)0, acc[j], false, false);

        buf ^= 1;
    }

    // epilogue: C/D layout — VGPR g holds rows g / g+8 (lane halves)
#pragma unroll
    for (int j = 0; j < 4; ++j) {
        const int col = col0 + j * 16 + l16;
        const float bv = bias ? bias[col] : 0.0f;
#pragma unroll
        for (int g = 0; g < 8; ++g) {
            const int row = row0 + wave * 16 + half * 8 + g;
            C[(size_t)row * N + col] = acc[j][g] + bv;
        }
    }
}

// ---------------------------------------------------------------------------
// Small two-part WMMA GEMM (LSTM gates, classifier):
//   C = act(A1 @ W1^T + A2 @ W2^T + bias1 + bias2), bf16 staged in LDS.
// ---------------------------------------------------------------------------
template <int BM>
__global__ __launch_bounds__(BM * 2) void wmma_gemm_kernel(
    const float* __restrict__ A1, const float* __restrict__ W1, int K1, int ldw1,
    const float* __restrict__ A2, const float* __restrict__ W2, int K2, int ldw2,
    const float* __restrict__ bias1, const float* __restrict__ bias2,
    float* __restrict__ C, int N, int act)
{
    constexpr int BN = 64;
    constexpr int NT = BM * 2;
    __shared__ __bf16 As[BM][32];
    __shared__ __bf16 Bs[BN][32];

    const int tid  = threadIdx.x;
    const int wave = tid >> 5;
    const int half = (tid >> 4) & 1;
    const int l16  = tid & 15;

    const int row0 = blockIdx.x * BM;
    const int col0 = blockIdx.y * BN;

    v8f acc[4];
#pragma unroll
    for (int j = 0; j < 4; ++j)
#pragma unroll
        for (int g = 0; g < 8; ++g) acc[j][g] = 0.0f;

#pragma unroll 1
    for (int part = 0; part < 2; ++part) {
        const float* A   = part ? A2 : A1;
        const float* W   = part ? W2 : W1;
        const int    K   = part ? K2 : K1;
        const int    ldw = part ? ldw2 : ldw1;
        if (K == 0) continue;

        for (int k0 = 0; k0 < K; k0 += 32) {
            __syncthreads();
            for (int i = tid; i < BM * 32; i += NT) {
                int r = i >> 5, k = i & 31;
                As[r][k] = (__bf16)A[(size_t)(row0 + r) * K + k0 + k];
            }
            for (int i = tid; i < BN * 32; i += NT) {
                int n = i >> 5, k = i & 31;
                Bs[n][k] = (__bf16)W[(size_t)(col0 + n) * ldw + k0 + k];
            }
            __syncthreads();

            v16bf a;
            {
                const __bf16* ar = &As[wave * 16 + l16][0];
#pragma unroll
                for (int e = 0; e < 8; ++e) {
                    a[e]     = ar[half * 16 + e];
                    a[e + 8] = ar[half * 16 + 8 + e];
                }
            }
            v16bf bf[4];
#pragma unroll
            for (int j = 0; j < 4; ++j) {
                const __bf16* br = &Bs[j * 16 + l16][0];
#pragma unroll
                for (int e = 0; e < 16; ++e) bf[j][e] = br[half * 16 + e];
            }
#pragma unroll
            for (int j = 0; j < 4; ++j)
                acc[j] = __builtin_amdgcn_wmma_f32_16x16x32_bf16(
                    false, a, false, bf[j], (short)0, acc[j], false, false);
        }
    }

#pragma unroll
    for (int j = 0; j < 4; ++j) {
        const int col = col0 + j * 16 + l16;
        float bsum = 0.0f;
        if (bias1) bsum += bias1[col];
        if (bias2) bsum += bias2[col];
#pragma unroll
        for (int g = 0; g < 8; ++g) {
            const int row = row0 + wave * 16 + half * 8 + g;
            float v = acc[j][g] + bsum;
            if (act) v = fmaxf(v, 0.0f);
            C[(size_t)row * N + col] = v;
        }
    }
}

// ---------------------------------------------------------------------------
// Elementwise helpers
// ---------------------------------------------------------------------------
__global__ void fill_f32(float* p, int n, float v) {
    int i = blockIdx.x * blockDim.x + threadIdx.x;
    if (i < n) p[i] = v;
}
__global__ void fill_u32(unsigned* p, int n, unsigned v) {
    int i = blockIdx.x * blockDim.x + threadIdx.x;
    if (i < n) p[i] = v;
}

__device__ __forceinline__ float sigf(float x) { return 1.0f / (1.0f + expf(-x)); }

__global__ void lstm_update(const float* __restrict__ gates,
                            float* __restrict__ h, float* __restrict__ c)
{
    int idx = blockIdx.x * blockDim.x + threadIdx.x;
    if (idx >= NB_GRAPHS * HIDDEN) return;
    int b = idx >> 9, j = idx & (HIDDEN - 1);
    const float* g = gates + (size_t)b * 4 * HIDDEN;
    float ig = sigf(g[j]);
    float fg = sigf(g[HIDDEN + j]);
    float gg = tanhf(g[2 * HIDDEN + j]);
    float og = sigf(g[3 * HIDDEN + j]);
    float cn = fg * c[idx] + ig * gg;
    c[idx] = cn;
    h[idx] = og * tanhf(cn);
}

// Ordered-uint encoding: float max == uint max (handles negatives).
__device__ __forceinline__ unsigned encKey(float x) {
    unsigned u = __float_as_uint(x);
    return (u & 0x80000000u) ? ~u : (u | 0x80000000u);
}
__device__ __forceinline__ float decKey(unsigned k) {
    unsigned u = (k & 0x80000000u) ? (k ^ 0x80000000u) : ~k;
    return __uint_as_float(u);
}

// pass1: e[n] = dot(x[n], h[batch[n]]); segment max via ordered-key atomic.
__global__ void attn_pass1(const float* __restrict__ x, const int* __restrict__ batch,
                           const float* __restrict__ h, float* __restrict__ e,
                           unsigned* __restrict__ mkey, int N)
{
    int node = (blockIdx.x * blockDim.x + threadIdx.x) >> 5;
    int lane = threadIdx.x & 31;
    if (node >= N) return;
    int b = batch[node];
    const float* xr = x + (size_t)node * HIDDEN;
    const float* hr = h + (size_t)b * HIDDEN;
    float s = 0.0f;
#pragma unroll
    for (int t = 0; t < HIDDEN / 32; ++t)
        s += xr[lane + 32 * t] * hr[lane + 32 * t];
#pragma unroll
    for (int m = 16; m >= 1; m >>= 1) s += __shfl_xor(s, m, 32);
    if (lane == 0) {
        e[node] = s;
        atomicMax(&mkey[b], encKey(s));
    }
}

// pass2: a = exp(e-m); s[b] += a; r[b,:] += a*x[n,:] (unnormalized),
// run-length batched atomics (batch ids are sorted).
__global__ void attn_pass2(const float* __restrict__ x, const int* __restrict__ batch,
                           const float* __restrict__ e, const unsigned* __restrict__ mkey,
                           float* __restrict__ s, float* __restrict__ r, int N)
{
    __shared__ float aS[256];
    __shared__ int   bS[256];
    const int n0  = blockIdx.x * 256;
    const int cnt = min(256, N - n0);
    const int tid = threadIdx.x;
    if (tid < cnt) {
        int n = n0 + tid;
        int b = batch[n];
        float m = decKey(mkey[b]);
        if (!isfinite(m)) m = 0.0f;
        float a = expf(e[n] - m);
        aS[tid] = a;
        bS[tid] = b;
        atomicAdd(&s[b], a);
    }
    __syncthreads();
    for (int ccol = tid; ccol < HIDDEN; ccol += 256) {
        float acc = 0.0f;
        int curb = bS[0];
        for (int i = 0; i < cnt; ++i) {
            int b = bS[i];
            if (b != curb) {
                atomicAdd(&r[(size_t)curb * HIDDEN + ccol], acc);
                acc = 0.0f;
                curb = b;
            }
            acc += aS[i] * x[(size_t)(n0 + i) * HIDDEN + ccol];
        }
        atomicAdd(&r[(size_t)curb * HIDDEN + ccol], acc);
    }
}

// q_star = [h | r/s]
__global__ void build_qstar(const float* __restrict__ h, const float* __restrict__ r,
                            const float* __restrict__ s, float* __restrict__ q)
{
    int idx = blockIdx.x * blockDim.x + threadIdx.x;
    if (idx >= NB_GRAPHS * 2 * HIDDEN) return;
    int b = idx >> 10, j = idx & (2 * HIDDEN - 1);
    float v;
    if (j < HIDDEN) {
        v = h[b * HIDDEN + j];
    } else {
        float sv = s[b];
        v = (sv != 0.0f) ? r[b * HIDDEN + (j - HIDDEN)] / sv : 0.0f;
    }
    q[idx] = v;
}

// out[b,j] = hdn[b,:] @ W2[j,:] + b2[j]  (64 x 2)
__global__ void out_kernel(const float* __restrict__ hdn, const float* __restrict__ W2,
                           const float* __restrict__ b2, float* __restrict__ out)
{
    int t = blockIdx.x * blockDim.x + threadIdx.x;
    if (t >= NB_GRAPHS * 2) return;
    int b = t >> 1, j = t & 1;
    float acc = b2[j];
    for (int k = 0; k < HIDDEN; ++k)
        acc += hdn[b * HIDDEN + k] * W2[j * HIDDEN + k];
    out[t] = acc;
}

// ---------------------------------------------------------------------------
// Host-side orchestration
// ---------------------------------------------------------------------------
static void run_set2set(const float* x, const int* batch, int N,
                        const float* Wih, const float* Whh,
                        const float* bih, const float* bhh,
                        float* h, float* c, float* qs, float* gates,
                        float* ebuf, unsigned* mkey, float* sbuf, float* rbuf,
                        hipStream_t stream)
{
    fill_f32<<<(NB_GRAPHS * HIDDEN + 255) / 256, 256, 0, stream>>>(h, NB_GRAPHS * HIDDEN, 0.0f);
    fill_f32<<<(NB_GRAPHS * HIDDEN + 255) / 256, 256, 0, stream>>>(c, NB_GRAPHS * HIDDEN, 0.0f);
    fill_f32<<<(NB_GRAPHS * 2 * HIDDEN + 255) / 256, 256, 0, stream>>>(qs, NB_GRAPHS * 2 * HIDDEN, 0.0f);

    for (int step = 0; step < 3; ++step) {
        wmma_gemm_kernel<64><<<dim3(1, (4 * HIDDEN) / 64), 128, 0, stream>>>(
            qs, Wih, 2 * HIDDEN, 2 * HIDDEN,
            h,  Whh, HIDDEN,     HIDDEN,
            bih, bhh, gates, 4 * HIDDEN, 0);
        lstm_update<<<(NB_GRAPHS * HIDDEN + 255) / 256, 256, 0, stream>>>(gates, h, c);

        fill_u32<<<1, 64, 0, stream>>>(mkey, NB_GRAPHS, 0x007FFFFFu);  // encKey(-inf)
        fill_f32<<<1, 64, 0, stream>>>(sbuf, NB_GRAPHS, 0.0f);
        fill_f32<<<(NB_GRAPHS * HIDDEN + 255) / 256, 256, 0, stream>>>(rbuf, NB_GRAPHS * HIDDEN, 0.0f);

        attn_pass1<<<(N + 7) / 8, 256, 0, stream>>>(x, batch, h, ebuf, mkey, N);
        attn_pass2<<<(N + 255) / 256, 256, 0, stream>>>(x, batch, ebuf, mkey, sbuf, rbuf, N);
        build_qstar<<<(NB_GRAPHS * 2 * HIDDEN + 255) / 256, 256, 0, stream>>>(h, rbuf, sbuf, qs);
    }
}

extern "C" void kernel_launch(void* const* d_in, const int* in_sizes, int n_in,
                              void* d_out, int out_size, void* d_ws, size_t ws_size,
                              hipStream_t stream)
{
    (void)in_sizes; (void)n_in; (void)out_size; (void)ws_size;

    const float* x_protein = (const float*)d_in[0];   // [32768, 1024]
    const int*   batch_p   = (const int*)d_in[1];
    const float* x_ligand  = (const float*)d_in[2];   // [2560, 128]
    const int*   batch_l   = (const int*)d_in[3];
    const float* Wp   = (const float*)d_in[4];
    const float* bp   = (const float*)d_in[5];
    const float* Wl   = (const float*)d_in[6];
    const float* bl   = (const float*)d_in[7];
    const float* pWih = (const float*)d_in[8];
    const float* pWhh = (const float*)d_in[9];
    const float* pbih = (const float*)d_in[10];
    const float* pbhh = (const float*)d_in[11];
    const float* lWih = (const float*)d_in[12];
    const float* lWhh = (const float*)d_in[13];
    const float* lbih = (const float*)d_in[14];
    const float* lbhh = (const float*)d_in[15];
    const float* W1   = (const float*)d_in[16];       // [512, 2048]
    const float* b1   = (const float*)d_in[17];
    const float* W2   = (const float*)d_in[18];       // [2, 512]
    const float* b2   = (const float*)d_in[19];

    const int NP = 32768, PDIM = 1024;
    const int NL = 2560,  ADIM = 128;

    float* ws    = (float*)d_ws;
    float* xp    = ws;
    float* xl    = xp + (size_t)NP * HIDDEN;
    float* gates = xl + (size_t)NL * HIDDEN;
    float* h     = gates + NB_GRAPHS * 4 * HIDDEN;
    float* c     = h + NB_GRAPHS * HIDDEN;
    float* qsp   = c + NB_GRAPHS * HIDDEN;
    float* qsl   = qsp + NB_GRAPHS * 2 * HIDDEN;
    float* ebuf  = qsl + NB_GRAPHS * 2 * HIDDEN;
    float* sbuf  = ebuf + NP;
    unsigned* mkey = (unsigned*)(sbuf + NB_GRAPHS);
    float* rbuf  = (float*)(mkey + NB_GRAPHS);
    float* hdn   = rbuf + NB_GRAPHS * HIDDEN;

    // xp = x_protein @ Wp^T + bp   [32768, 512]  (TDM-staged)
    wmma_gemm_tdm_kernel<<<dim3(NP / 128, HIDDEN / 64), 256, 0, stream>>>(
        x_protein, Wp, PDIM, PDIM, bp, xp, HIDDEN);

    // xl = x_ligand @ Wl^T + bl    [2560, 512]   (TDM-staged)
    wmma_gemm_tdm_kernel<<<dim3(NL / 128, HIDDEN / 64), 256, 0, stream>>>(
        x_ligand, Wl, ADIM, ADIM, bl, xl, HIDDEN);

    run_set2set(xp, batch_p, NP, pWih, pWhh, pbih, pbhh,
                h, c, qsp, gates, ebuf, mkey, sbuf, rbuf, stream);
    run_set2set(xl, batch_l, NL, lWih, lWhh, lbih, lbhh,
                h, c, qsl, gates, ebuf, mkey, sbuf, rbuf, stream);

    // hdn = relu([qsp | qsl] @ W1^T + b1)   [64, 512]
    wmma_gemm_kernel<64><<<dim3(1, HIDDEN / 64), 128, 0, stream>>>(
        qsp, W1,              2 * HIDDEN, 4 * HIDDEN,
        qsl, W1 + 2 * HIDDEN, 2 * HIDDEN, 4 * HIDDEN,
        b1, nullptr, hdn, HIDDEN, 1);

    out_kernel<<<1, 128, 0, stream>>>(hdn, W2, b2, (float*)d_out);
}
